// DualModal_ASSM_20332375179632
// MI455X (gfx1250) — matrix operand order repaired and checked by reading, not compiled, not run on hardware
//
#include <hip/hip_runtime.h>
#include <math.h>

// ---------------------------------------------------------------------------
// MI455X (gfx1250) implementation notes:
//  * wave32 only; all matrix math via v_wmma_f32_16x16x32_f16 (f32 accum).
//  * Activations kept token-major [N, C] in f16 so WMMA A (weights, [M,K]
//    row-major) and B (X, [N,K] row-major) fragments are contiguous 16B loads
//    matching the CDNA5 16-bit A/B VGPR layouts (CDNA5 ISA).
//  * Each wave computes a 32x32 output tile (2 m-tiles x 2 n-tiles): 4 WMMAs
//    per 8 b128 loads -> 2 loads/WMMA, halving redundant fragment traffic.
//  * Problem is HBM-bound (~1-2GB traffic @ 23.3TB/s >> 70 GFLOP @ f16 WMMA),
//    so f16 storage halves traffic and WMMA keeps compute off the roofline.
// ---------------------------------------------------------------------------

typedef _Float16 h16;
typedef __attribute__((ext_vector_type(16))) _Float16 v16h;
typedef __attribute__((ext_vector_type(8)))  float    v8f;
typedef __attribute__((ext_vector_type(4)))  float    f32x4;

#define NTOK 65536   // B*H*W
#define HW   16384
#define NB   4
#define DH   192     // hidden = 2*DIM
#define CIN  96      // DIM

__device__ __forceinline__ float geluf(float x) {
  return 0.5f * x * (1.0f + erff(x * 0.70710678118654752f));
}
__device__ __forceinline__ float sigm(float x) {
  return 1.0f / (1.0f + expf(-x));
}

// ---------- input transpose/convert: [B,C,HW] f32 -> [N,C] f16 (LDS tiled) --
__global__ void k_cvt_inputs(const float* __restrict__ ms, const float* __restrict__ pan,
                             h16* __restrict__ xms, h16* __restrict__ xpan,
                             h16* __restrict__ xf) {
  __shared__ float tms[32][33];
  __shared__ float tpan[32][33];
  int b  = blockIdx.z;
  int c0 = blockIdx.y * 32;
  int l0 = blockIdx.x * 32;
  int tx = threadIdx.x, ty = threadIdx.y;
  size_t base = (size_t)b * CIN * HW;
  for (int i = 0; i < 4; ++i) {
    int c = c0 + ty + i * 8;
    tms [ty + i * 8][tx] = ms [base + (size_t)c * HW + l0 + tx];
    tpan[ty + i * 8][tx] = pan[base + (size_t)c * HW + l0 + tx];
  }
  __syncthreads();
  for (int i = 0; i < 4; ++i) {
    int l = l0 + ty + i * 8;
    int c = c0 + tx;
    size_t o = ((size_t)b * HW + l) * CIN + c;
    float a = tms[tx][ty + i * 8];
    float p = tpan[tx][ty + i * 8];
    xms[o]  = (h16)a;
    xpan[o] = (h16)p;
    xf[o]   = (h16)((a + p) * 0.5f);
  }
}

// ---------- weight f32 -> f16 --------------------------------------------
__global__ void k_cvt_w(const float* __restrict__ s, h16* __restrict__ d, int n) {
  int i = blockIdx.x * 256 + threadIdx.x;
  if (i < n) d[i] = (h16)s[i];
}

// ---------- WMMA GEMM epilogue helper -------------------------------------
__device__ __forceinline__ void epi_store(v8f c, int n, int mb,
                                          const float* __restrict__ bias,
                                          int act, int mode,
                                          h16* __restrict__ Y, int ldY,
                                          float* __restrict__ Yf, int M) {
  float r[8];
#pragma unroll
  for (int i = 0; i < 8; ++i) {
    float v = c[i] + bias[mb + i];
    if (act == 1) v = geluf(v);
    else if (act == 2) v = sigm(v);
    r[i] = v;
  }
  if (mode == 0) {
    union { f32x4 q; h16 h[8]; } o;
#pragma unroll
    for (int i = 0; i < 8; ++i) o.h[i] = (h16)r[i];
    *(f32x4*)(Y + (size_t)n * ldY + mb) = o.q;
  } else {
    int bb = n >> 14, l = n & (HW - 1);
    size_t ob = (size_t)bb * M * HW + l;
#pragma unroll
    for (int i = 0; i < 8; ++i)
      Yf[ob + (size_t)(mb + i) * HW] = r[i];
  }
}

// ---------- generic WMMA GEMM: Y[n,m] = act(sum_k W[m,k]*X[n,k] + bias[m]) --
// split-B: k < K0 reads X0 (ld0), k >= K0 reads X1 (ld1)  (zero-copy concat)
// Each wave: 32x32 output tile = 2 m-tiles x 2 n-tiles, 4 WMMA per k-step.
// mode 0: store f16 token-major Y[n*ldY + m]
// mode 1: store f32 channel-major Yf[b*M*HW + m*HW + l]   (final heads)
__global__ __launch_bounds__(256)
void k_gemm(const h16* __restrict__ W, const float* __restrict__ bias,
            const h16* __restrict__ X0, int ld0, int K0,
            const h16* __restrict__ X1, int ld1,
            h16* __restrict__ Y, int ldY,
            float* __restrict__ Yf,
            int M, int K, int act, int mode) {
  int wave = threadIdx.x >> 5;
  int lane = threadIdx.x & 31;
  int n0 = (blockIdx.x * 8 + wave) << 5;   // two n-tiles: n0, n0+16
  int m0 = blockIdx.y << 5;                // two m-tiles: m0, m0+16
  int tm = lane & 15, kh = lane >> 4;

  union F { v16h v; f32x4 q[2]; };
  F a0, a1, b0, b1;
  v8f c00 = {}, c01 = {}, c10 = {}, c11 = {};

  const h16* wr0 = W + (size_t)(m0 + tm) * K + kh * 8;
  const h16* wr1 = wr0 + (size_t)16 * K;
  int kb = 0;
#pragma unroll 1
  for (int pass = 0; pass < 2; ++pass) {
    const h16* Xp; int ldp, kend, kofs;
    if (pass == 0) { Xp = X0; ldp = ld0; kend = (K0 < K ? K0 : K); kofs = 0; }
    else           { Xp = X1; ldp = ld1; kend = K;                 kofs = K0; }
    if (kb >= kend) continue;
    const h16* xr0 = Xp + (size_t)(n0 + tm) * ldp + kh * 8 - kofs;
    const h16* xr1 = xr0 + (size_t)16 * ldp;
    for (; kb < kend; kb += 32) {
      // A 16x32 f16 fragments: two contiguous 16B chunks per lane
      a0.q[0] = *(const f32x4*)(wr0 + kb);
      a0.q[1] = *(const f32x4*)(wr0 + kb + 16);
      a1.q[0] = *(const f32x4*)(wr1 + kb);
      a1.q[1] = *(const f32x4*)(wr1 + kb + 16);
      // B 32x16 f16 fragments: two contiguous 16B chunks per lane (token-major X)
      b0.q[0] = *(const f32x4*)(xr0 + kb);
      b0.q[1] = *(const f32x4*)(xr0 + kb + 16);
      b1.q[0] = *(const f32x4*)(xr1 + kb);
      b1.q[1] = *(const f32x4*)(xr1 + kb + 16);
      __builtin_prefetch(xr0 + kb + 32, 0, 3);  // global_prefetch_b8
      __builtin_prefetch(xr1 + kb + 32, 0, 3);
      c00 = __builtin_amdgcn_wmma_f32_16x16x32_f16(false, a0.v, false, b0.v,
                                                   (short)0, c00, false, false);
      c01 = __builtin_amdgcn_wmma_f32_16x16x32_f16(false, a0.v, false, b1.v,
                                                   (short)0, c01, false, false);
      c10 = __builtin_amdgcn_wmma_f32_16x16x32_f16(false, a1.v, false, b0.v,
                                                   (short)0, c10, false, false);
      c11 = __builtin_amdgcn_wmma_f32_16x16x32_f16(false, a1.v, false, b1.v,
                                                   (short)0, c11, false, false);
    }
  }
  int na = n0 + tm, nb2 = n0 + 16 + tm;
  int ma = m0 + kh * 8, mb2 = m0 + 16 + kh * 8;
  epi_store(c00, na,  ma,  bias, act, mode, Y, ldY, Yf, M);
  epi_store(c01, nb2, ma,  bias, act, mode, Y, ldY, Yf, M);
  epi_store(c10, na,  mb2, bias, act, mode, Y, ldY, Yf, M);
  epi_store(c11, nb2, mb2, bias, act, mode, Y, ldY, Yf, M);
}

// ---------- depthwise conv (3x3 or 5x5), token-major ----------------------
// mode 0: y = conv(x)+b        mode 1 (CPE): y = x * sigmoid(conv(x)+b)
__global__ void k_dwconv(const h16* __restrict__ x, const float* __restrict__ w,
                         const float* __restrict__ bias, h16* __restrict__ y,
                         int ks, int mode) {
  int idx = blockIdx.x * 256 + threadIdx.x;  // over NTOK*DH
  int c = idx % DH;
  int n = idx / DH;
  int l = n & (HW - 1);
  int py = l >> 7, px = l & 127;
  int r = ks >> 1;
  float acc = bias[c];
  const float* wc = w + c * ks * ks;
  for (int dy = 0; dy < ks; ++dy) {
    int yy = py + dy - r;
    if ((unsigned)yy >= 128u) continue;
    for (int dx = 0; dx < ks; ++dx) {
      int xx = px + dx - r;
      if ((unsigned)xx >= 128u) continue;
      int nn = n + (dy - r) * 128 + (dx - r);
      acc += (float)x[(size_t)nn * DH + c] * wc[dy * ks + dx];
    }
  }
  if (mode == 0) y[idx] = (h16)acc;
  else           y[idx] = (h16)((float)x[idx] * sigm(acc));
}

// ---------- SE pooling: deterministic 2-stage reduction -------------------
__global__ void k_pool(const h16* __restrict__ x, float* __restrict__ part) {
  int b = blockIdx.y, ch = blockIdx.x, c = threadIdx.x;  // block = 192 thr
  float s = 0.f;
  size_t base = ((size_t)b * HW + ch * 256) * DH + c;
  for (int t = 0; t < 256; ++t) s += (float)x[base + (size_t)t * DH];
  part[((size_t)b * 64 + ch) * DH + c] = s;
}

__global__ void k_fe(const float* __restrict__ part,
                     const float* __restrict__ fe1w, const float* __restrict__ fe1b,
                     const float* __restrict__ fe2w, const float* __restrict__ fe2b,
                     float* __restrict__ scale) {
  __shared__ float m[DH];
  __shared__ float g[48];
  int b = blockIdx.x, c = threadIdx.x;  // block = 192 thr
  float s = 0.f;
  for (int k = 0; k < 64; ++k) s += part[((size_t)b * 64 + k) * DH + c];
  m[c] = s * (1.0f / HW);
  __syncthreads();
  if (c < 48) {
    float a = fe1b[c];
    for (int k = 0; k < DH; ++k) a += fe1w[c * DH + k] * m[k];
    g[c] = geluf(a);
  }
  __syncthreads();
  float a = fe2b[c];
  for (int k = 0; k < 48; ++k) a += fe2w[c * 48 + k] * g[k];
  scale[b * DH + c] = 1.0f + sigm(a);  // reference: x * (1 + sigmoid(.))
}

// ---------- routing: argmax(logits + gumbel) -> prompt gather -------------
// policy == one_hot(argmax) in forward; lse shift cancels in argmax.
__global__ void k_route(const h16* __restrict__ logits,
                        const float* __restrict__ msemb, const float* __restrict__ panemb,
                        const float* __restrict__ tokw, const float* __restrict__ tokb,
                        float* __restrict__ pms, float* __restrict__ ppan) {
  int n = blockIdx.x * 256 + threadIdx.x;
  const h16* lg = logits + (size_t)n * 64;
  float best = -1e30f; int bj = 0;
  for (int j = 0; j < 64; ++j) {
    // deterministic splitmix64 gumbel surrogate (JAX threefry not reproduced)
    unsigned long long h = (((unsigned long long)n << 6) | (unsigned)j) + 42ull * 0x9E3779B97F4A7C15ull;
    h = (h ^ (h >> 30)) * 0xBF58476D1CE4E5B9ull;
    h = (h ^ (h >> 27)) * 0x94D049BB133111EBull;
    h ^= h >> 31;
    float u = ((float)(unsigned)(h >> 40) + 0.5f) * (1.0f / 16777216.0f);
    u = fminf(fmaxf(u, 1e-7f), 1.0f - 1e-7f);
    float gmb = -logf(-logf(u));
    float v = (float)lg[j] + gmb;
    if (v > best) { best = v; bj = j; }
  }
  const float* ems = msemb + bj * 32;
  const float* epn = panemb + bj * 32;
  for (int s = 0; s < 16; ++s) {
    float a = tokb[s], p = tokb[s];
    for (int r = 0; r < 32; ++r) {
      float tw = tokw[s * 32 + r];
      a += tw * ems[r];
      p += tw * epn[r];
    }
    pms [(size_t)n * 16 + s] = a;
    ppan[(size_t)n * 16 + s] = p;
  }
}

// ---------- layernorm over C=192, one wave32 per token --------------------
// optional prompt add: pp[n][c%16] (the jnp.tile by rf=12)
__global__ void k_ln(const h16* __restrict__ x, const float* __restrict__ pp,
                     const float* __restrict__ gam, const float* __restrict__ bet,
                     h16* __restrict__ y) {
  int lane = threadIdx.x & 31;
  int n = blockIdx.x * 8 + (threadIdx.x >> 5);
  const h16* xr = x + (size_t)n * DH;
  float v[6], s1 = 0.f, s2 = 0.f;
#pragma unroll
  for (int k = 0; k < 6; ++k) {
    int c = lane + k * 32;
    float t = (float)xr[c];
    if (pp) t += pp[(size_t)n * 16 + (c & 15)];
    v[k] = t; s1 += t; s2 += t * t;
  }
#pragma unroll
  for (int o = 16; o > 0; o >>= 1) {
    s1 += __shfl_xor(s1, o, 32);
    s2 += __shfl_xor(s2, o, 32);
  }
  float mean = s1 * (1.0f / DH);
  float var  = s2 * (1.0f / DH) - mean * mean;
  float rs = rsqrtf(var + 1e-5f);
  h16* yr = y + (size_t)n * DH;
#pragma unroll
  for (int k = 0; k < 6; ++k) {
    int c = lane + k * 32;
    yr[c] = (h16)((v[k] - mean) * rs * gam[c] + bet[c]);
  }
}

// ---------- elementwise fusion kernels ------------------------------------
// ms_h = ms_h*(1+freq) + ms_local*g_ms*0.7 ; same for pan (gate = [N,384])
__global__ void k_combine1(h16* __restrict__ msh, h16* __restrict__ panh,
                           const h16* __restrict__ msl, const h16* __restrict__ panl,
                           const h16* __restrict__ gate,
                           const float* __restrict__ sms, const float* __restrict__ spn) {
  int idx = blockIdx.x * 256 + threadIdx.x;
  int c = idx % DH, n = idx / DH, b = n >> 14;
  float gms = (float)gate[(size_t)n * 384 + c];
  float gpn = (float)gate[(size_t)n * 384 + 192 + c];
  msh[idx]  = (h16)((float)msh[idx]  * sms[b * DH + c] + (float)msl[idx]  * gms * 0.7f);
  panh[idx] = (h16)((float)panh[idx] * spn[b * DH + c] + (float)panl[idx] * gpn * 0.7f);
}

__global__ void k_mul(h16* __restrict__ a, const h16* __restrict__ b) {
  int i = blockIdx.x * 256 + threadIdx.x;
  a[i] = (h16)((float)a[i] * (float)b[i]);
}

// e = e + f*g*0.6 + r*ls[c]
__global__ void k_ocomb(h16* __restrict__ e, const h16* __restrict__ f,
                        const h16* __restrict__ g, const h16* __restrict__ r,
                        const float* __restrict__ ls) {
  int i = blockIdx.x * 256 + threadIdx.x;
  int c = i % DH;
  e[i] = (h16)((float)e[i] + (float)f[i] * (float)g[i] * 0.6f + (float)r[i] * ls[c]);
}

// ===========================================================================
extern "C" void kernel_launch(void* const* d_in, const int* in_sizes, int n_in,
                              void* d_out, int out_size, void* d_ws, size_t ws_size,
                              hipStream_t stream) {
  (void)in_sizes; (void)out_size; (void)ws_size;
  const float* IN[64];
  for (int i = 0; i < 64 && i < n_in; ++i) IN[i] = (const float*)d_in[i];

  char* ws = (char*)d_ws;
  size_t off = 0;
  auto alloc = [&](size_t bytes) -> char* {
    char* p = ws + off;
    off += (bytes + 255) & ~(size_t)255;
    return p;
  };
  auto cvt = [&](int idx, int n) -> h16* {
    h16* d = (h16*)alloc((size_t)n * 2);
    k_cvt_w<<<dim3((n + 255) / 256), 256, 0, stream>>>(IN[idx], d, n);
    return d;
  };
  auto gemm = [&](const h16* W, const float* bias,
                  const h16* X0, int ld0, int K0, const h16* X1, int ld1,
                  h16* Y, int ldY, float* Yf, int M, int K, int act, int mode) {
    dim3 g(NTOK / 256, M / 32);
    k_gemm<<<g, 256, 0, stream>>>(W, bias, X0, ld0, K0, X1, ld1, Y, ldY, Yf, M, K, act, mode);
  };

  // ---- weight conversions (f32 -> f16, ~2MB total) ----
  h16* w_msproj = cvt(2,  DH * CIN);
  h16* w_pnproj = cvt(4,  DH * CIN);
  h16* w_lf1    = cvt(12, DH * 2 * DH);
  h16* w_lf2    = cvt(14, DH * DH);
  h16* w_mg     = cvt(20, 2 * DH * 2 * DH);
  h16* w_r1     = cvt(22, 32 * CIN);
  h16* w_r2     = cvt(24, 64 * 32);
  h16* w_msl1   = cvt(30, DH * DH);
  h16* w_msl2   = cvt(32, DH * DH);
  h16* w_pnl1   = cvt(34, DH * DH);
  h16* w_pnl2   = cvt(36, DH * DH);
  h16* w_f1     = cvt(38, 2 * DH * 2 * DH);
  h16* w_f2     = cvt(40, DH * 2 * DH);
  h16* w_msg    = cvt(48, DH * 2 * DH);
  h16* w_png    = cvt(50, DH * 2 * DH);
  h16* w_ctx    = cvt(52, DH * 2 * DH);
  h16* w_msres  = cvt(54, DH * DH);
  h16* w_pnres  = cvt(56, DH * DH);
  h16* w_out    = cvt(62, CIN * DH);

  // ---- activation buffers ([N,192] f16 = 24MB each; lifetimes reused) ----
  size_t SZ = (size_t)NTOK * DH * 2;
  h16* H1 = (h16*)alloc(SZ);                       // ms_h (post-cpe .. ms_sem)
  h16* H2 = (h16*)alloc(SZ);                       // pan_h
  h16* S1 = (h16*)alloc(SZ);                       // scratch (logits/a3/maps/res)
  h16* S2 = (h16*)alloc(SZ);                       // scratch (a5/fusion)
  h16* S3 = (h16*)alloc(SZ);                       // scratch (gelu tmp/gates)
  h16* L1 = (h16*)alloc(SZ);                       // x_ms -> ms_local
  h16* L2 = (h16*)alloc(SZ);                       // x_pan -> pan_local
  h16* E1 = (h16*)alloc(SZ);                       // x_fused -> ms_enh -> ms_o
  h16* E2 = (h16*)alloc(SZ);                       // r1out  -> pan_enh -> pan_o
  h16* Wd = (h16*)alloc((size_t)NTOK * 2 * DH * 2); // [N,384]: mg gate / f1 out
  float* pms   = (float*)alloc((size_t)NTOK * 16 * 4);
  float* ppan  = (float*)alloc((size_t)NTOK * 16 * 4);
  float* partm = (float*)alloc((size_t)NB * 64 * DH * 4);
  float* partp = (float*)alloc((size_t)NB * 64 * DH * 4);
  float* scm   = (float*)alloc((size_t)NB * DH * 4);
  float* scp   = (float*)alloc((size_t)NB * DH * 4);

  h16* x_ms = L1; h16* x_pan = L2; h16* x_fused = E1; h16* r1out = E2; h16* logits = S1;
  const int BIG = 1 << 30;
  dim3 EW(NTOK * DH / 256);

  // 1) inputs -> f16 token-major (+ fused = (ms+pan)/2)
  k_cvt_inputs<<<dim3(HW / 32, CIN / 32, NB), dim3(32, 8), 0, stream>>>(
      IN[0], IN[1], x_ms, x_pan, x_fused);

  // 2) routing: r1(gelu) -> r2 -> argmax(+gumbel) -> prompt gather
  gemm(w_r1, IN[23], x_fused, CIN, BIG, nullptr, 0, r1out, 32, nullptr, 32, CIN, 1, 0);
  gemm(w_r2, IN[25], r1out, 32, BIG, nullptr, 0, logits, 64, nullptr, 64, 32, 0, 0);
  k_route<<<dim3(NTOK / 256), 256, 0, stream>>>(logits, IN[26], IN[27], IN[28], IN[29], pms, ppan);

  // 3) 1x1 projections + CPE sigmoid gating
  gemm(w_msproj, IN[3], x_ms, CIN, BIG, nullptr, 0, S1, DH, nullptr, DH, CIN, 0, 0);
  k_dwconv<<<EW, 256, 0, stream>>>(S1, IN[6], IN[7], H1, 3, 1);
  gemm(w_pnproj, IN[5], x_pan, CIN, BIG, nullptr, 0, S1, DH, nullptr, DH, CIN, 0, 0);
  k_dwconv<<<EW, 256, 0, stream>>>(S1, IN[6], IN[7], H2, 3, 1);

  // 4) local enhancement: dw3 + dw5 -> lf1(gelu, split-B concat) -> lf2
  k_dwconv<<<EW, 256, 0, stream>>>(H1, IN[8], IN[9], S1, 3, 0);
  k_dwconv<<<EW, 256, 0, stream>>>(H1, IN[10], IN[11], S2, 5, 0);
  gemm(w_lf1, IN[13], S1, DH, DH, S2, DH, S3, DH, nullptr, DH, 2 * DH, 1, 0);
  gemm(w_lf2, IN[15], S3, DH, BIG, nullptr, 0, L1, DH, nullptr, DH, DH, 0, 0);
  k_dwconv<<<EW, 256, 0, stream>>>(H2, IN[8], IN[9], S1, 3, 0);
  k_dwconv<<<EW, 256, 0, stream>>>(H2, IN[10], IN[11], S2, 5, 0);
  gemm(w_lf1, IN[13], S1, DH, DH, S2, DH, S3, DH, nullptr, DH, 2 * DH, 1, 0);
  gemm(w_lf2, IN[15], S3, DH, BIG, nullptr, 0, L2, DH, nullptr, DH, DH, 0, 0);

  // 5) SE frequency gates (deterministic 2-stage reduce)
  k_pool<<<dim3(64, NB), 192, 0, stream>>>(H1, partm);
  k_pool<<<dim3(64, NB), 192, 0, stream>>>(H2, partp);
  k_fe<<<dim3(NB), 192, 0, stream>>>(partm, IN[16], IN[17], IN[18], IN[19], scm);
  k_fe<<<dim3(NB), 192, 0, stream>>>(partp, IN[16], IN[17], IN[18], IN[19], scp);

  // 6) modality gate (sigmoid) over concat(locals) + combine
  gemm(w_mg, IN[21], L1, DH, DH, L2, DH, Wd, 2 * DH, nullptr, 2 * DH, 2 * DH, 2, 0);
  k_combine1<<<EW, 256, 0, stream>>>(H1, H2, L1, L2, Wd, scm, scp);

  // 7) per-modality map + LN(+tiled prompt)   [sort/unsort is identity]
  gemm(w_msl1, IN[31], H1, DH, BIG, nullptr, 0, S3, DH, nullptr, DH, DH, 1, 0);
  gemm(w_msl2, IN[33], S3, DH, BIG, nullptr, 0, S1, DH, nullptr, DH, DH, 0, 0);
  k_ln<<<dim3(NTOK / 8), 256, 0, stream>>>(S1, pms, IN[42], IN[43], E1);
  gemm(w_pnl1, IN[35], H2, DH, BIG, nullptr, 0, S3, DH, nullptr, DH, DH, 1, 0);
  gemm(w_pnl2, IN[37], S3, DH, BIG, nullptr, 0, S1, DH, nullptr, DH, DH, 0, 0);
  k_ln<<<dim3(NTOK / 8), 256, 0, stream>>>(S1, ppan, IN[44], IN[45], E2);

  // 8) cross fusion: f1(gelu, split-B) -> f2 -> LN
  gemm(w_f1, IN[39], E1, DH, DH, E2, DH, Wd, 2 * DH, nullptr, 2 * DH, 2 * DH, 1, 0);
  gemm(w_f2, IN[41], Wd, 2 * DH, BIG, nullptr, 0, S1, DH, nullptr, DH, 2 * DH, 0, 0);
  k_ln<<<dim3(NTOK / 8), 256, 0, stream>>>(S1, nullptr, IN[46], IN[47], S2);

  // 9) gates (sigmoid, split-B over [ms_sem|pan_sem]) + residuals
  gemm(w_ctx, IN[53], H1, DH, DH, H2, DH, S3, DH, nullptr, DH, 2 * DH, 2, 0);
  k_mul<<<EW, 256, 0, stream>>>(S2, S3);  // fusion_g = fusion * ctx_gate
  gemm(w_msg, IN[49], H1, DH, DH, H2, DH, S3, DH, nullptr, DH, 2 * DH, 2, 0);
  gemm(w_msres, IN[55], H1, DH, BIG, nullptr, 0, S1, DH, nullptr, DH, DH, 0, 0);
  k_ocomb<<<EW, 256, 0, stream>>>(E1, S2, S3, S1, IN[58]);
  gemm(w_png, IN[51], H1, DH, DH, H2, DH, S3, DH, nullptr, DH, 2 * DH, 2, 0);
  gemm(w_pnres, IN[57], H2, DH, BIG, nullptr, 0, S1, DH, nullptr, DH, DH, 0, 0);
  k_ocomb<<<EW, 256, 0, stream>>>(E2, S2, S3, S1, IN[59]);

  // 10) output heads: LN -> out GEMM, stored f32 channel-major into d_out
  float* out = (float*)d_out;
  k_ln<<<dim3(NTOK / 8), 256, 0, stream>>>(E1, nullptr, IN[60], IN[61], S1);
  gemm(w_out, IN[63], S1, DH, BIG, nullptr, 0, nullptr, 0, out, CIN, DH, 0, 1);
  k_ln<<<dim3(NTOK / 8), 256, 0, stream>>>(E2, nullptr, IN[60], IN[61], S1);
  gemm(w_out, IN[63], S1, DH, BIG, nullptr, 0, nullptr, 0,
       out + (size_t)NB * CIN * HW, CIN, DH, 0, 1);
}
